// NeuralGraphSimple_63763084476611
// MI455X (gfx1250) — matrix operand order repaired and checked by reading, not compiled, not run on hardware
//
#include <hip/hip_runtime.h>

// ---------------------------------------------------------------------------
// Model dimensions (compile-time, from the reference)
// ---------------------------------------------------------------------------
#define BATCH   1024
#define CHAN    22
#define TLEN    1675
#define NEDGE   132          // C * K = 22 * 6
#define NNODES  (BATCH * CHAN)

typedef __attribute__((ext_vector_type(16))) __bf16 v16bf;
typedef __attribute__((ext_vector_type(8)))  __bf16 v8bf;
typedef __attribute__((ext_vector_type(8)))  float  v8f;

// float -> bf16 round-to-nearest-even via integer ops
__device__ __forceinline__ __bf16 f2bf(float f) {
    unsigned u = __builtin_bit_cast(unsigned, f);
    unsigned r = (u + 0x7FFFu + ((u >> 16) & 1u)) >> 16;
    unsigned short h = (unsigned short)r;
    return __builtin_bit_cast(__bf16, h);
}

__device__ __forceinline__ v16bf join16(v8bf lo, v8bf hi) {
    return __builtin_shufflevector(lo, hi, 0, 1, 2, 3, 4, 5, 6, 7,
                                   8, 9, 10, 11, 12, 13, 14, 15);
}

// ---------------------------------------------------------------------------
// Weight packer: B panels [Kpad/32][N][32] bf16 (per 32-K panel, each column n
// holds its 32 K-elements contiguously -> B fragment = 2x 16B loads).
//   transposed=1: w is [N, Kdim] row-major (conv weights [Cout, Cin*Kw])
//   transposed=0: w is [Kdim, N] row-major (dense layer weights)
// ---------------------------------------------------------------------------
__global__ __launch_bounds__(256) void pack_weights(const float* __restrict__ w,
                                                    __bf16* __restrict__ Bp,
                                                    int Kdim, int Kpad, int N,
                                                    int transposed) {
    int idx = blockIdx.x * blockDim.x + threadIdx.x;
    if (idx >= Kpad * N) return;
    int k = idx / N, n = idx - k * N;
    float v = 0.0f;
    if (k < Kdim) v = transposed ? w[(size_t)n * Kdim + k] : w[(size_t)k * N + n];
    Bp[((size_t)(k >> 5) * N + n) * 32 + (k & 31)] = f2bf(v);
}

// Convert f32 [M, Kdim] -> bf16 [M, Kpad] (zero pad K tail)
__global__ __launch_bounds__(256) void convert_pad_bf16(const float* __restrict__ src,
                                                        __bf16* __restrict__ dst,
                                                        long M, int Kdim, int Kpad) {
    long i = (long)blockIdx.x * blockDim.x + threadIdx.x;
    if (i >= M * Kpad) return;
    long m = i / Kpad;
    int  k = (int)(i - m * Kpad);
    float v = (k < Kdim) ? src[m * Kdim + k] : 0.0f;
    dst[i] = f2bf(v);
}

// ---------------------------------------------------------------------------
// Plain bf16 GEMM: out[M,N] = A[M,Kpad](bf16) x B(panels) + bias
// Block = 128 threads = 4 waves; each wave owns one 16-row M tile and NT
// 16-col N tiles (NT = N/16, template).  M must be a multiple of 64.
// Per k-step: one A fragment (2x b128), then ALL NT B fragments issued as one
// load clause, then an NT-deep back-to-back WMMA chain (disjoint accumulators,
// no D->A/B hazards).
// ---------------------------------------------------------------------------
template <int NT>
__global__ __launch_bounds__(128) void gemm_bf16(const __bf16* __restrict__ A,
                                                 const __bf16* __restrict__ Bp,
                                                 const float* __restrict__ bias,
                                                 float* __restrict__ out,
                                                 int Kpad) {
    const int N     = NT * 16;
    const int lane  = threadIdx.x & 31;
    const int wm    = threadIdx.x >> 5;          // wave's M tile in block
    const int mrow  = lane & 15;
    const int khalf = lane >> 4;
    const int ncol  = lane & 15;
    const int m0    = blockIdx.x * 64 + wm * 16;

    const __bf16* arow = A + (size_t)(m0 + mrow) * Kpad;

    v8f acc[NT];
#pragma unroll
    for (int t = 0; t < NT; ++t) acc[t] = (v8f){};

    for (int k0 = 0; k0 < Kpad; k0 += 32) {
        if (k0 + 128 < Kpad)
            __builtin_prefetch(arow + k0 + 128, 0, 1);     // global_prefetch_b8

        // A fragment: elements 0..7 = K khalf*8..+7 ; 8..15 = K 16+khalf*8..+7
        const __bf16* ap = arow + k0 + khalf * 8;
        v16bf a = join16(*(const v8bf*)ap, *(const v8bf*)(ap + 16));

        // issue all B fragments first (one clause, one wait)
        const __bf16* bpanel = Bp + (size_t)(k0 >> 5) * N * 32;
        v16bf bfr[NT];
#pragma unroll
        for (int t = 0; t < NT; ++t) {
            const __bf16* bp = bpanel + (size_t)(t * 16 + ncol) * 32 + khalf * 8;
            bfr[t] = join16(*(const v8bf*)bp, *(const v8bf*)(bp + 16));
        }
        // back-to-back WMMA chain
#pragma unroll
        for (int t = 0; t < NT; ++t) {
            acc[t] = __builtin_amdgcn_wmma_f32_16x16x32_bf16(
                false, a, false, bfr[t], (short)0, acc[t], false, false);
        }
    }

#pragma unroll
    for (int t = 0; t < NT; ++t) {
        int n = t * 16 + ncol;
        float bv = bias ? bias[n] : 0.0f;
#pragma unroll
        for (int r = 0; r < 8; ++r) {
            int mm = m0 + r + (khalf ? 8 : 0);
            out[(size_t)mm * N + n] = acc[t][r] + bv;
        }
    }
}

// ---------------------------------------------------------------------------
// Conv1d as WMMA GEMM with LDS-staged im2col window.
// Block = 128 threads = 4 waves; handles one batch sample and 64 consecutive
// output positions (4 M tiles); each wave owns one M tile x all NT N tiles.
// LDS: input window [(Cin+1) x Lwin] bf16 (last row zeros for K padding) and
// an offset LUT k -> cin*Lwin + kk (kills div/mod in the inner loop).
// ---------------------------------------------------------------------------
template <int NT>
__global__ __launch_bounds__(128) void conv_wmma(const float* __restrict__ x,
                                                 const __bf16* __restrict__ Bp,
                                                 const float* __restrict__ bias,
                                                 float* __restrict__ out,
                                                 int Cin, int Lin, int Lout,
                                                 int Kw, int stride, int pad,
                                                 int Kdim, int Kpad, int Lwin) {
    __shared__ __bf16 sIn[9216];      // max (128+1)*66 = 8514 used
    __shared__ int    saoff[384];     // max Kpad = 384
    const int N   = NT * 16;
    const int tid = threadIdx.x;
    const int b   = blockIdx.y;
    const int lo0 = blockIdx.x * 64;
    const int winstart = lo0 * stride - pad;
    const float* xb = x + (size_t)b * Cin * Lin;

    // stage input window (+ zero row) with padding/bounds resolved here
    const int nstage = (Cin + 1) * Lwin;
    for (int idx = tid; idx < nstage; idx += 128) {
        int cin = idx / Lwin, p = idx - cin * Lwin;
        int pos = winstart + p;
        float v = 0.0f;
        if (cin < Cin && pos >= 0 && pos < Lin) v = xb[(size_t)cin * Lin + pos];
        sIn[idx] = f2bf(v);
    }
    // stage k -> window-offset LUT (padding K tail -> zero row)
    for (int k = tid; k < Kpad; k += 128) {
        int cin = k / Kw, kk = k - cin * Kw;
        saoff[k] = (k < Kdim) ? (cin * Lwin + kk) : (Cin * Lwin);
    }
    __syncthreads();

    const int lane  = tid & 31;
    const int wm    = tid >> 5;
    const int mrow  = lane & 15;
    const int khalf = lane >> 4;
    const int ncol  = lane & 15;
    const int mbase = (wm * 16 + mrow) * stride;   // lane's window offset

    v8f acc[NT];
#pragma unroll
    for (int t = 0; t < NT; ++t) acc[t] = (v8f){};

    for (int k0 = 0; k0 < Kpad; k0 += 32) {
        // gather A fragment from LDS through the offset LUT
        v16bf a;
#pragma unroll
        for (int i = 0; i < 8; ++i) {
            int kb = (i < 4) ? (khalf * 8 + 2 * i) : (16 + khalf * 8 + 2 * (i - 4));
            int k  = k0 + kb;
            a[2 * i]     = sIn[saoff[k]     + mbase];
            a[2 * i + 1] = sIn[saoff[k + 1] + mbase];
        }
        // issue all B fragments first, then the WMMA chain
        const __bf16* bpanel = Bp + (size_t)(k0 >> 5) * N * 32;
        v16bf bfr[NT];
#pragma unroll
        for (int t = 0; t < NT; ++t) {
            const __bf16* bp = bpanel + (size_t)(t * 16 + ncol) * 32 + khalf * 8;
            bfr[t] = join16(*(const v8bf*)bp, *(const v8bf*)(bp + 16));
        }
#pragma unroll
        for (int t = 0; t < NT; ++t) {
            acc[t] = __builtin_amdgcn_wmma_f32_16x16x32_bf16(
                false, a, false, bfr[t], (short)0, acc[t], false, false);
        }
    }

#pragma unroll
    for (int t = 0; t < NT; ++t) {
        int n = t * 16 + ncol;
        float bv = bias[n];
#pragma unroll
        for (int r = 0; r < 8; ++r) {
            int lo = lo0 + wm * 16 + r + (khalf ? 8 : 0);
            if (lo < Lout)
                out[((size_t)b * N + n) * Lout + lo] = acc[t][r] + bv;
        }
    }
}

// ---------------------------------------------------------------------------
// BatchNorm stats over [B, C, L] per channel (sum, sumsq); one block / channel
// ---------------------------------------------------------------------------
__global__ __launch_bounds__(256) void bn_stats_nch(const float* __restrict__ x,
                                                    int Bn, int Cc, int L,
                                                    float* __restrict__ sum,
                                                    float* __restrict__ sumsq) {
    __shared__ float s1[256], s2[256];
    const int c = blockIdx.x, tid = threadIdx.x;
    float a = 0.f, b = 0.f;
    const long cnt = (long)Bn * L;
    for (long i = tid; i < cnt; i += 256) {
        long bb = i / L, l = i - bb * L;
        float v = x[((size_t)bb * Cc + c) * L + l];
        a += v; b += v * v;
    }
    s1[tid] = a; s2[tid] = b; __syncthreads();
    for (int s = 128; s > 0; s >>= 1) {
        if (tid < s) { s1[tid] += s1[tid + s]; s2[tid] += s2[tid + s]; }
        __syncthreads();
    }
    if (tid == 0) { sum[c] = s1[0]; sumsq[c] = s2[0]; }
}

__global__ __launch_bounds__(256) void bn_apply_relu_nch(const float* __restrict__ x,
                                                         float* __restrict__ y,
                                                         const float* __restrict__ gam,
                                                         const float* __restrict__ bet,
                                                         const float* __restrict__ sum,
                                                         const float* __restrict__ sumsq,
                                                         int Bn, int Cc, int L) {
    long i = (long)blockIdx.x * blockDim.x + threadIdx.x;
    if (i >= (long)Bn * Cc * L) return;
    int c = (int)((i / L) % Cc);
    float cnt = (float)Bn * (float)L;
    float m = sum[c] / cnt;
    float v = sumsq[c] / cnt - m * m;
    float sc = gam[c] * rsqrtf(v + 1e-5f);
    y[i] = fmaxf((x[i] - m) * sc + bet[c], 0.0f);
}

// BatchNorm over [N, F] per feature
__global__ __launch_bounds__(256) void bn_stats_nc(const float* __restrict__ x,
                                                   int Nr, int F,
                                                   float* __restrict__ sum,
                                                   float* __restrict__ sumsq) {
    __shared__ float s1[256], s2[256];
    const int f = blockIdx.x, tid = threadIdx.x;
    float a = 0.f, b = 0.f;
    for (int i = tid; i < Nr; i += 256) {
        float v = x[(size_t)i * F + f];
        a += v; b += v * v;
    }
    s1[tid] = a; s2[tid] = b; __syncthreads();
    for (int s = 128; s > 0; s >>= 1) {
        if (tid < s) { s1[tid] += s1[tid + s]; s2[tid] += s2[tid + s]; }
        __syncthreads();
    }
    if (tid == 0) { sum[f] = s1[0]; sumsq[f] = s2[0]; }
}

__global__ __launch_bounds__(256) void bn_apply_relu_nc(const float* __restrict__ x,
                                                        float* __restrict__ y,
                                                        const float* __restrict__ gam,
                                                        const float* __restrict__ bet,
                                                        const float* __restrict__ sum,
                                                        const float* __restrict__ sumsq,
                                                        int Nr, int F) {
    long i = (long)blockIdx.x * blockDim.x + threadIdx.x;
    if (i >= (long)Nr * F) return;
    int f = (int)(i % F);
    float m = sum[f] / (float)Nr;
    float v = sumsq[f] / (float)Nr - m * m;
    float sc = gam[f] * rsqrtf(v + 1e-5f);
    y[i] = fmaxf((x[i] - m) * sc + bet[f], 0.0f);
}

// MaxPool1d(kernel=2, stride=2) over [BC rows, Lin]
__global__ __launch_bounds__(256) void maxpool2(const float* __restrict__ x,
                                                float* __restrict__ y,
                                                long BC, int Lin) {
    int Lout = Lin >> 1;
    long i = (long)blockIdx.x * blockDim.x + threadIdx.x;
    if (i >= BC * Lout) return;
    long r = i / Lout; int l = (int)(i - r * Lout);
    const float* xr = x + r * Lin + 2 * l;
    y[i] = fmaxf(xr[0], xr[1]);
}

// mean over last dim: [rows, L] -> [rows]
__global__ __launch_bounds__(256) void mean_last(const float* __restrict__ x,
                                                 float* __restrict__ y,
                                                 long rows, int L) {
    long r = (long)blockIdx.x * blockDim.x + threadIdx.x;
    if (r >= rows) return;
    float s = 0.f;
    for (int l = 0; l < L; ++l) s += x[r * L + l];
    y[r] = s / (float)L;
}

// Graph degree / symmetric normalization (single block)
__global__ void gcn_norm(const int* __restrict__ ei,
                         float* __restrict__ enorm, float* __restrict__ dinv2) {
    __shared__ int deg[CHAN];
    __shared__ float sdinv[CHAN];
    int tid = threadIdx.x;
    if (tid < CHAN) deg[tid] = 1;               // self loop
    __syncthreads();
    if (tid < NEDGE) atomicAdd(&deg[ei[NEDGE + tid]], 1);
    __syncthreads();
    if (tid < CHAN) {
        float d = rsqrtf((float)deg[tid]);
        sdinv[tid] = d;
        dinv2[tid] = d * d;
    }
    __syncthreads();
    if (tid < NEDGE) enorm[tid] = sdinv[ei[tid]] * sdinv[ei[NEDGE + tid]];
}

// GCN aggregate: per node, per feature; edges identical per sample graph
__global__ __launch_bounds__(64) void gcn_aggregate(const float* __restrict__ h,
                                                    float* __restrict__ out,
                                                    const int* __restrict__ ei,
                                                    const float* __restrict__ enorm,
                                                    const float* __restrict__ dinv2,
                                                    const float* __restrict__ bias,
                                                    int F) {
    __shared__ int   ssrc[NEDGE], sdst[NEDGE];
    __shared__ float sn[NEDGE];
    const int node = blockIdx.x;
    const int f = threadIdx.x;
    for (int e = f; e < NEDGE; e += F) {
        ssrc[e] = ei[e]; sdst[e] = ei[NEDGE + e]; sn[e] = enorm[e];
    }
    __syncthreads();
    const int c = node % CHAN;
    const int base = node - c;
    float acc = h[(size_t)node * F + f] * dinv2[c];
    for (int e = 0; e < NEDGE; ++e)
        if (sdst[e] == c) acc += h[(size_t)(base + ssrc[e]) * F + f] * sn[e];
    out[(size_t)node * F + f] = acc + bias[f];
}

// mean over 22 nodes: [B, 22, F] -> [B, F]
__global__ __launch_bounds__(256) void mean_nodes(const float* __restrict__ g2,
                                                  float* __restrict__ gf, int F) {
    int i = blockIdx.x * blockDim.x + threadIdx.x;
    if (i >= BATCH * F) return;
    int b = i / F, f = i - b * F;
    float s = 0.f;
    for (int c = 0; c < CHAN; ++c) s += g2[((size_t)b * CHAN + c) * F + f];
    gf[i] = s / (float)CHAN;
}

// concat [B,Fa] ++ [B,Fb] -> [B,Fa+Fb]
__global__ __launch_bounds__(256) void concat2(const float* __restrict__ a, int Fa,
                                               const float* __restrict__ b, int Fb,
                                               float* __restrict__ out) {
    int F = Fa + Fb;
    int i = blockIdx.x * blockDim.x + threadIdx.x;
    if (i >= BATCH * F) return;
    int bi = i / F, f = i - bi * F;
    out[i] = (f < Fa) ? a[(size_t)bi * Fa + f] : b[(size_t)bi * Fb + (f - Fa)];
}

// final 64->2 linear + softmax (too small for WMMA)
__global__ __launch_bounds__(256) void fc3_softmax(const float* __restrict__ h,
                                                   const float* __restrict__ w,
                                                   const float* __restrict__ b2,
                                                   float* __restrict__ logits,
                                                   float* __restrict__ probs) {
    int bi = blockIdx.x * blockDim.x + threadIdx.x;
    if (bi >= BATCH) return;
    float z0 = b2[0], z1 = b2[1];
    for (int k = 0; k < 64; ++k) {
        float v = h[(size_t)bi * 64 + k];
        z0 += v * w[k * 2 + 0];
        z1 += v * w[k * 2 + 1];
    }
    logits[bi * 2] = z0; logits[bi * 2 + 1] = z1;
    float mx = fmaxf(z0, z1);
    float e0 = expf(z0 - mx), e1 = expf(z1 - mx);
    float s = e0 + e1;
    probs[bi * 2] = e0 / s; probs[bi * 2 + 1] = e1 / s;
}

// ---------------------------------------------------------------------------
// Host-side orchestration
// ---------------------------------------------------------------------------
extern "C" void kernel_launch(void* const* d_in, const int* in_sizes, int n_in,
                              void* d_out, int out_size, void* d_ws, size_t ws_size,
                              hipStream_t stream) {
    (void)in_sizes; (void)n_in; (void)out_size; (void)ws_size;

    const float* x   = (const float*)d_in[0];   // [1024, 22, 1675]
    const int*   ei  = (const int*)d_in[1];     // [2, 132]
    const float* c1w = (const float*)d_in[2],  *c1b  = (const float*)d_in[3];
    const float* bn1g= (const float*)d_in[4],  *bn1b = (const float*)d_in[5];
    const float* c2w = (const float*)d_in[6],  *c2b  = (const float*)d_in[7];
    const float* bn2g= (const float*)d_in[8],  *bn2b = (const float*)d_in[9];
    const float* c3w = (const float*)d_in[10], *c3b  = (const float*)d_in[11];
    const float* bn3g= (const float*)d_in[12], *bn3b = (const float*)d_in[13];
    const float* c4w = (const float*)d_in[14], *c4b  = (const float*)d_in[15];
    const float* bn4g= (const float*)d_in[16], *bn4b = (const float*)d_in[17];
    const float* g1w = (const float*)d_in[18], *g1b  = (const float*)d_in[19];
    const float* gbn1g=(const float*)d_in[20], *gbn1b= (const float*)d_in[21];
    const float* g2w = (const float*)d_in[22], *g2b  = (const float*)d_in[23];
    const float* gbn2g=(const float*)d_in[24], *gbn2b= (const float*)d_in[25];
    const float* l1w = (const float*)d_in[26], *l1b  = (const float*)d_in[27];
    const float* cbn1g=(const float*)d_in[28], *cbn1b= (const float*)d_in[29];
    const float* l2w = (const float*)d_in[30], *l2b  = (const float*)d_in[31];
    const float* cbn2g=(const float*)d_in[32], *cbn2b= (const float*)d_in[33];
    const float* l3w = (const float*)d_in[34], *l3b  = (const float*)d_in[35];

    // ---- workspace carve-up --------------------------------------------
    char* ws = (char*)d_ws;
    size_t off = 0;
    auto take = [&](size_t bytes) -> char* {
        char* p = ws + off;
        off += (bytes + 255) & ~(size_t)255;
        return p;
    };
    // bufA doubles as xbf (GCN runs before CNN): 110 MB >= 76.4 MB
    float*  actA = (float*)take((size_t)BATCH * 32 * 838 * 4);
    float*  actB = (float*)take((size_t)BATCH * 32 * 419 * 4);
    __bf16* xbf  = (__bf16*)actA;                       // [22528, 1696] bf16
    __bf16* wp1 = (__bf16*)take((size_t)160  * 32  * 2);
    __bf16* wp2 = (__bf16*)take((size_t)160  * 64  * 2);
    __bf16* wp3 = (__bf16*)take((size_t)192  * 128 * 2);
    __bf16* wp4 = (__bf16*)take((size_t)384  * 128 * 2);
    __bf16* g1p = (__bf16*)take((size_t)1696 * 64  * 2);
    __bf16* g2p = (__bf16*)take((size_t)64   * 64  * 2);
    __bf16* l1p = (__bf16*)take((size_t)192  * 128 * 2);
    __bf16* l2p = (__bf16*)take((size_t)128  * 64  * 2);
    float* ssum     = (float*)take(128 * 4);
    float* ssq      = (float*)take(128 * 4);
    float* cnn_feat = (float*)take((size_t)BATCH * 128 * 4);
    float* h1       = (float*)take((size_t)NNODES * 64 * 4);
    float* a1       = (float*)take((size_t)NNODES * 64 * 4);
    float* h2       = (float*)take((size_t)NNODES * 64 * 4);
    float* a2       = (float*)take((size_t)NNODES * 64 * 4);
    __bf16* abf     = (__bf16*)take((size_t)NNODES * 64 * 2);
    float* gfeat    = (float*)take((size_t)BATCH * 64 * 4);
    float* fc1      = (float*)take((size_t)BATCH * 128 * 4);
    float* fc2      = (float*)take((size_t)BATCH * 64 * 4);
    __bf16* cbf     = (__bf16*)take((size_t)BATCH * 192 * 2);
    __bf16* f1bf    = (__bf16*)take((size_t)BATCH * 128 * 2);
    float* dinv2    = (float*)take(CHAN * 4);
    float* enorm    = (float*)take(NEDGE * 4);

    float* logits   = (float*)d_out;                 // [1024, 2]
    float* probs    = (float*)d_out + 2048;          // [1024, 2]
    float* combined = (float*)d_out + 4096;          // [1024, 192]

    auto nblk = [](long n, int t) { return (unsigned)((n + t - 1) / t); };
    auto pack = [&](const float* w, __bf16* Bp, int Kdim, int Kpad, int N, int tr) {
        pack_weights<<<nblk((long)Kpad * N, 256), 256, 0, stream>>>(w, Bp, Kdim, Kpad, N, tr);
    };

    // ---- pack all weights into bf16 K-panel layout ----------------------
    pack(c1w, wp1, 22 * 7,  160,  32, 1);
    pack(c2w, wp2, 32 * 5,  160,  64, 1);
    pack(c3w, wp3, 64 * 3,  192, 128, 1);
    pack(c4w, wp4, 128 * 3, 384, 128, 1);
    pack(g1w, g1p, TLEN,   1696,  64, 0);
    pack(g2w, g2p, 64,       64,  64, 0);
    pack(l1w, l1p, 192,     192, 128, 0);
    pack(l2w, l2p, 128,     128,  64, 0);

    // ---- GCN branch (first: xbf aliases the CNN ping buffer) ------------
    gcn_norm<<<1, 160, 0, stream>>>(ei, enorm, dinv2);

    convert_pad_bf16<<<nblk((long)NNODES * 1696, 256), 256, 0, stream>>>(
        x, xbf, NNODES, TLEN, 1696);
    gemm_bf16<4><<<NNODES / 64, 128, 0, stream>>>(xbf, g1p, nullptr, h1, 1696);
    gcn_aggregate<<<NNODES, 64, 0, stream>>>(h1, a1, ei, enorm, dinv2, g1b, 64);
    bn_stats_nc<<<64, 256, 0, stream>>>(a1, NNODES, 64, ssum, ssq);
    bn_apply_relu_nc<<<nblk((long)NNODES * 64, 256), 256, 0, stream>>>(
        a1, a1, gbn1g, gbn1b, ssum, ssq, NNODES, 64);

    convert_pad_bf16<<<nblk((long)NNODES * 64, 256), 256, 0, stream>>>(
        a1, abf, NNODES, 64, 64);
    gemm_bf16<4><<<NNODES / 64, 128, 0, stream>>>(abf, g2p, nullptr, h2, 64);
    gcn_aggregate<<<NNODES, 64, 0, stream>>>(h2, a2, ei, enorm, dinv2, g2b, 64);
    bn_stats_nc<<<64, 256, 0, stream>>>(a2, NNODES, 64, ssum, ssq);
    bn_apply_relu_nc<<<nblk((long)NNODES * 64, 256), 256, 0, stream>>>(
        a2, a2, gbn2g, gbn2b, ssum, ssq, NNODES, 64);
    mean_nodes<<<nblk((long)BATCH * 64, 256), 256, 0, stream>>>(a2, gfeat, 64);

    // ---- CNN branch -----------------------------------------------------
    // conv1: [1024,22,1675] -> [1024,32,838]   (Lwin = 63*2+7 = 133)
    conv_wmma<2><<<dim3(nblk(838, 64), BATCH), 128, 0, stream>>>(
        x, wp1, c1b, actA, 22, 1675, 838, 7, 2, 3, 154, 160, 133);
    bn_stats_nch<<<32, 256, 0, stream>>>(actA, BATCH, 32, 838, ssum, ssq);
    bn_apply_relu_nch<<<nblk((long)BATCH * 32 * 838, 256), 256, 0, stream>>>(
        actA, actA, bn1g, bn1b, ssum, ssq, BATCH, 32, 838);
    maxpool2<<<nblk((long)BATCH * 32 * 419, 256), 256, 0, stream>>>(
        actA, actB, (long)BATCH * 32, 838);

    // conv2: [1024,32,419] -> [1024,64,210]    (Lwin = 63*2+5 = 131)
    conv_wmma<4><<<dim3(nblk(210, 64), BATCH), 128, 0, stream>>>(
        actB, wp2, c2b, actA, 32, 419, 210, 5, 2, 2, 160, 160, 131);
    bn_stats_nch<<<64, 256, 0, stream>>>(actA, BATCH, 64, 210, ssum, ssq);
    bn_apply_relu_nch<<<nblk((long)BATCH * 64 * 210, 256), 256, 0, stream>>>(
        actA, actA, bn2g, bn2b, ssum, ssq, BATCH, 64, 210);
    maxpool2<<<nblk((long)BATCH * 64 * 105, 256), 256, 0, stream>>>(
        actA, actB, (long)BATCH * 64, 210);

    // conv3: [1024,64,105] -> [1024,128,53]    (Lwin = 63*2+3 = 129)
    conv_wmma<8><<<dim3(1, BATCH), 128, 0, stream>>>(
        actB, wp3, c3b, actA, 64, 105, 53, 3, 2, 1, 192, 192, 129);
    bn_stats_nch<<<128, 256, 0, stream>>>(actA, BATCH, 128, 53, ssum, ssq);
    bn_apply_relu_nch<<<nblk((long)BATCH * 128 * 53, 256), 256, 0, stream>>>(
        actA, actA, bn3g, bn3b, ssum, ssq, BATCH, 128, 53);
    maxpool2<<<nblk((long)BATCH * 128 * 26, 256), 256, 0, stream>>>(
        actA, actB, (long)BATCH * 128, 53);

    // conv4: [1024,128,26] -> [1024,128,26]    (Lwin = 63*1+3 = 66)
    conv_wmma<8><<<dim3(1, BATCH), 128, 0, stream>>>(
        actB, wp4, c4b, actA, 128, 26, 26, 3, 1, 1, 384, 384, 66);
    bn_stats_nch<<<128, 256, 0, stream>>>(actA, BATCH, 128, 26, ssum, ssq);
    bn_apply_relu_nch<<<nblk((long)BATCH * 128 * 26, 256), 256, 0, stream>>>(
        actA, actA, bn4g, bn4b, ssum, ssq, BATCH, 128, 26);
    mean_last<<<nblk((long)BATCH * 128, 256), 256, 0, stream>>>(
        actA, cnn_feat, (long)BATCH * 128, 26);

    // ---- head -----------------------------------------------------------
    concat2<<<nblk((long)BATCH * 192, 256), 256, 0, stream>>>(
        cnn_feat, 128, gfeat, 64, combined);

    convert_pad_bf16<<<nblk((long)BATCH * 192, 256), 256, 0, stream>>>(
        combined, cbf, BATCH, 192, 192);
    gemm_bf16<8><<<BATCH / 64, 128, 0, stream>>>(cbf, l1p, l1b, fc1, 192);
    bn_stats_nc<<<128, 256, 0, stream>>>(fc1, BATCH, 128, ssum, ssq);
    bn_apply_relu_nc<<<nblk((long)BATCH * 128, 256), 256, 0, stream>>>(
        fc1, fc1, cbn1g, cbn1b, ssum, ssq, BATCH, 128);

    convert_pad_bf16<<<nblk((long)BATCH * 128, 256), 256, 0, stream>>>(
        fc1, f1bf, BATCH, 128, 128);
    gemm_bf16<4><<<BATCH / 64, 128, 0, stream>>>(f1bf, l2p, l2b, fc2, 128);
    bn_stats_nc<<<64, 256, 0, stream>>>(fc2, BATCH, 64, ssum, ssq);
    bn_apply_relu_nc<<<nblk((long)BATCH * 64, 256), 256, 0, stream>>>(
        fc2, fc2, cbn2g, cbn2b, ssum, ssq, BATCH, 64);

    fc3_softmax<<<nblk(BATCH, 256), 256, 0, stream>>>(fc2, l3w, l3b, logits, probs);
}
